// CausalMultiHeadAttention_28011776704690
// MI455X (gfx1250) — matrix operand order repaired and checked by reading, not compiled, not run on hardware
//
#include <hip/hip_runtime.h>

// ---- problem constants (fixed shapes from the reference) ----
#define BATCH 4
#define SEQ   2048
#define DMODEL 1024
#define NHEAD 16
#define HDIM  64
#define MROWS (BATCH * SEQ)   // 8192

// 1/sqrt(HD) * log2(e): folded into Q so softmax can use raw exp2 (v_exp_f32).
#define QSCALE (0.125f * 1.44269504088896340736f)

typedef __bf16 v8bf  __attribute__((ext_vector_type(8)));
typedef __bf16 v16bf __attribute__((ext_vector_type(16)));
typedef float  v8f   __attribute__((ext_vector_type(8)));

#if __has_builtin(__builtin_amdgcn_exp2f)
#define EXP2F(x) __builtin_amdgcn_exp2f(x)
#else
#define EXP2F(x) exp2f(x)
#endif

// ---------------- WMMA helpers ----------------
__device__ __forceinline__ v8f wmma_bf16(v16bf a, v16bf b, v8f c) {
    // D = A(16x32 bf16) * B(32x16 bf16) + C(16x16 f32)
    return __builtin_amdgcn_wmma_f32_16x16x32_bf16(
        /*neg_a=*/false, a, /*neg_b=*/false, b,
        /*c_mod=*/(short)0, c, /*reuse_a=*/false, /*reuse_b=*/false);
}

__device__ __forceinline__ v16bf combine16(v8bf lo, v8bf hi) {
    v16bf r;
#pragma unroll
    for (int i = 0; i < 8; ++i) { r[i] = lo[i]; r[i + 8] = hi[i]; }
    return r;
}

// A-fragment: 16x32 bf16 tile, row-major with stride ld.
// Lane layout (ISA 7.12.2): row M = lane&15; K chunks [g*8, g*8+7] and [16+g*8, 16+g*8+7].
__device__ __forceinline__ v16bf ld_fragA(const __bf16* __restrict__ base, int ld) {
    const int lane = threadIdx.x & 31;
    const int r = lane & 15, g = lane >> 4;
    const __bf16* p = base + (size_t)r * ld + g * 8;
    v8bf lo = *(const v8bf*)(p);
    v8bf hi = *(const v8bf*)(p + 16);
    return combine16(lo, hi);
}

// B-fragment: 32x16 bf16 tile given a k-major operand Bt[n][k] (stride ld).
// Lane layout: col N = lane&15; K chunks [g*8, g*8+7] and [16+g*8, 16+g*8+7].
__device__ __forceinline__ v16bf ld_fragB(const __bf16* __restrict__ base, int ld) {
    const int lane = threadIdx.x & 31;
    const int n = lane & 15, g = lane >> 4;
    const __bf16* p = base + (size_t)n * ld + g * 8;
    v8bf lo = *(const v8bf*)(p);
    v8bf hi = *(const v8bf*)(p + 16);
    return combine16(lo, hi);
}

// ---- 16-lane-half butterfly reductions.
// DPP (VALU pipe, co-issues with WMMA) instead of ds_bpermute (LDS pipe).
// xor1 = quad_perm[1,0,3,2]=0xB1, xor2 = quad_perm[2,3,0,1]=0x4E,
// stage3 = row_half_mirror (0x141), stage4 = row_mirror (0x140).
#if __has_builtin(__builtin_amdgcn_update_dpp)
template <int CTRL>
__device__ __forceinline__ float dpp_xfer(float x) {
    int i = __float_as_int(x);
    int j = __builtin_amdgcn_update_dpp(i, i, CTRL, 0xF, 0xF, false);
    return __int_as_float(j);
}
__device__ __forceinline__ float half_max16(float x) {
    x = fmaxf(x, dpp_xfer<0xB1>(x));
    x = fmaxf(x, dpp_xfer<0x4E>(x));
    x = fmaxf(x, dpp_xfer<0x141>(x));
    x = fmaxf(x, dpp_xfer<0x140>(x));
    return x;
}
__device__ __forceinline__ float half_sum16(float x) {
    x += dpp_xfer<0xB1>(x);
    x += dpp_xfer<0x4E>(x);
    x += dpp_xfer<0x141>(x);
    x += dpp_xfer<0x140>(x);
    return x;
}
#else
__device__ __forceinline__ float half_max16(float x) {
    x = fmaxf(x, __shfl_xor(x, 1));
    x = fmaxf(x, __shfl_xor(x, 2));
    x = fmaxf(x, __shfl_xor(x, 4));
    x = fmaxf(x, __shfl_xor(x, 8));
    return x;
}
__device__ __forceinline__ float half_sum16(float x) {
    x += __shfl_xor(x, 1);
    x += __shfl_xor(x, 2);
    x += __shfl_xor(x, 4);
    x += __shfl_xor(x, 8);
    return x;
}
#endif

// ---------------- conversion kernels ----------------
__global__ void cvt_bf16_kernel(const float* __restrict__ in, __bf16* __restrict__ out, int n) {
    int i = blockIdx.x * blockDim.x + threadIdx.x;
    if (i < n) out[i] = (__bf16)in[i];
}

// Wt[n*D + k] = bf16(W[k*D + n])   (D = 1024)
__global__ void transpose_w_kernel(const float* __restrict__ W, __bf16* __restrict__ Wt) {
    int idx = blockIdx.x * blockDim.x + threadIdx.x;   // over D*D
    int n = idx >> 10;
    int k = idx & 1023;
    Wt[idx] = (__bf16)W[(size_t)k * DMODEL + n];
}

// ---------------- WMMA GEMM: C[M,N] = A[M,K] * Bt[N,K]^T ----------------
// MODE 0: store bf16 into [B,H,S,HD]            (K projection)
// MODE 1: store bf16*QSCALE into [B,H,S,HD]     (Q projection, scale folded)
// MODE 2: store bf16 into [B,H,HD,S]            (V transposed)
// MODE 3: store f32 row-major + bias            (final projection)
template <int MODE>
__global__ __launch_bounds__(128) void gemm_wmma_kernel(
    const __bf16* __restrict__ A, const __bf16* __restrict__ Bt,
    const float* __restrict__ bias, void* __restrict__ out)
{
    const int lane = threadIdx.x & 31;
    const int wave = threadIdx.x >> 5;
    const int r = lane & 15, g = lane >> 4;
    const int m0 = blockIdx.y * 64 + wave * 16;
    const int n0 = blockIdx.x * 64;

    v8f acc[4] = {};
    for (int k0 = 0; k0 < DMODEL; k0 += 32) {
        v16bf a = ld_fragA(A + (size_t)m0 * DMODEL + k0, DMODEL);
#pragma unroll
        for (int nn = 0; nn < 4; ++nn) {
            v16bf b = ld_fragB(Bt + (size_t)(n0 + nn * 16) * DMODEL + k0, DMODEL);
            acc[nn] = wmma_bf16(a, b, acc[nn]);
        }
    }

#pragma unroll
    for (int v = 0; v < 8; ++v) {
        const int m = m0 + v + g * 8;           // C layout: M = vgpr + 8*lanegroup
#pragma unroll
        for (int nn = 0; nn < 4; ++nn) {
            const int n = n0 + nn * 16 + r;     // C layout: N = lane&15
            float val = acc[nn][v];
            if (MODE == 3) {
                ((float*)out)[(size_t)m * DMODEL + n] = val + bias[n];
            } else {
                if (MODE == 1) val *= QSCALE;
                const int b_ = m >> 11, s = m & (SEQ - 1);
                const int h_ = n >> 6, hd = n & (HDIM - 1);
                if (MODE == 2) {  // [B,H,HD,S]
                    ((__bf16*)out)[(((size_t)(b_ * NHEAD + h_)) * HDIM + hd) * SEQ + s] = (__bf16)val;
                } else {          // [B,H,S,HD]
                    ((__bf16*)out)[(((size_t)(b_ * NHEAD + h_)) * SEQ + s) * HDIM + hd] = (__bf16)val;
                }
            }
        }
    }
}

// ---------------- flash attention tile step ----------------
// MASKED=true only for the single diagonal tile per wave; bulk tiles skip all
// causal compares/cndmasks.
template <bool MASKED>
__device__ __forceinline__ void kv_tile_step(
    const __bf16* __restrict__ Kb, const __bf16* __restrict__ Vb,
    __bf16 (*__restrict__ pw)[32],              // this wave's 16x32 LDS tile
    const v16bf& aq0, const v16bf& aq1,
    int kv0, int q0, int r, int g,
    float (&m_run)[8], float (&l_run)[8], v8f (&o)[4])
{
    // ---- scores: 16(q) x 32(kv), two 16x16 C tiles, K-dim = HD (2x32) ----
    v8f sc[2];
#pragma unroll
    for (int n = 0; n < 2; ++n) {
        v16bf b0 = ld_fragB(Kb + (size_t)(kv0 + n * 16) * HDIM + 0, HDIM);
        v16bf b1 = ld_fragB(Kb + (size_t)(kv0 + n * 16) * HDIM + 32, HDIM);
        v8f s = {};
        s = wmma_bf16(aq0, b0, s);
        s = wmma_bf16(aq1, b1, s);
        sc[n] = s;
    }

    // ---- online softmax in the (already log2e-scaled) logit domain ----
#pragma unroll
    for (int v = 0; v < 8; ++v) {
        float s0 = sc[0][v];
        float s1 = sc[1][v];
        if (MASKED) {
            const int qg = q0 + v + g * 8;
            if (kv0 + r > qg)      s0 = -3.0e38f;
            if (kv0 + 16 + r > qg) s1 = -3.0e38f;
        }
        float mx = half_max16(fmaxf(s0, s1));
        const float m_new = fmaxf(m_run[v], mx);
        const float corr = EXP2F(m_run[v] - m_new);
        const float p0 = EXP2F(s0 - m_new);
        const float p1 = EXP2F(s1 - m_new);
        l_run[v] = l_run[v] * corr + half_sum16(p0 + p1);
        m_run[v] = m_new;
#pragma unroll
        for (int nn = 0; nn < 4; ++nn) o[nn][v] *= corr;

        // scatter P: row M = v+g*8, cols r and 16+r (C-layout -> LDS tile)
        pw[v + g * 8][r]      = (__bf16)p0;
        pw[v + g * 8][16 + r] = (__bf16)p1;
    }

    // wave-private LDS dependency: stores -> (wait) -> A-layout reload
    asm volatile("s_wait_dscnt 0" ::: "memory");
    v8bf plo = *(const v8bf*)&pw[r][g * 8];
    v8bf phi = *(const v8bf*)&pw[r][16 + g * 8];
    const v16bf ap = combine16(plo, phi);

    // ---- O += P(16x32) * V(32x64) ----
#pragma unroll
    for (int nn = 0; nn < 4; ++nn) {
        v16bf bv = ld_fragB(Vb + (size_t)(nn * 16) * SEQ + kv0, SEQ);
        o[nn] = wmma_bf16(ap, bv, o[nn]);
    }
}

// ---------------- flash attention (causal, online softmax) ----------------
// grid: (SEQ/64, BATCH*NHEAD), block: 128 (4 waves). Wave handles 16 q rows x HD=64.
__global__ __launch_bounds__(128) void flash_attn_kernel(
    const __bf16* __restrict__ Q,   // [B,H,S,HD], pre-scaled by QSCALE
    const __bf16* __restrict__ K,   // [B,H,S,HD]
    const __bf16* __restrict__ Vt,  // [B,H,HD,S]
    __bf16* __restrict__ ctx)       // [B,S,D]  (D = H*HD, head-interleaved)
{
    __shared__ __bf16 plds[4][16][32];   // per-wave P staging tile

    const int lane = threadIdx.x & 31;
    const int wave = threadIdx.x >> 5;
    const int r = lane & 15, g = lane >> 4;
    const int bh = blockIdx.y;
    const int b = bh >> 4, h = bh & (NHEAD - 1);
    const int q0 = blockIdx.x * 64 + wave * 16;

    const __bf16* Qb = Q + (size_t)bh * SEQ * HDIM;
    const __bf16* Kb = K + (size_t)bh * SEQ * HDIM;
    const __bf16* Vb = Vt + (size_t)bh * HDIM * SEQ;

    // Q A-fragments, resident for the whole kv loop (K-dim = HD = 2 x 32)
    const v16bf aq0 = ld_fragA(Qb + (size_t)q0 * HDIM + 0, HDIM);
    const v16bf aq1 = ld_fragA(Qb + (size_t)q0 * HDIM + 32, HDIM);

    float m_run[8], l_run[8];
    v8f o[4] = {};
#pragma unroll
    for (int v = 0; v < 8; ++v) { m_run[v] = -3.0e38f; l_run[v] = 0.0f; }

    // Bulk: tiles fully below the causal diagonal (kv0+31 <= q0) -> no masking.
    const int kv_full_end = ((q0 + 1) >> 5) << 5;
    for (int kv0 = 0; kv0 < kv_full_end; kv0 += 32)
        kv_tile_step<false>(Kb, Vb, plds[wave], aq0, aq1, kv0, q0, r, g, m_run, l_run, o);
    // Exactly one diagonal tile per wave needs the causal mask.
    kv_tile_step<true>(Kb, Vb, plds[wave], aq0, aq1, kv_full_end, q0, r, g, m_run, l_run, o);

    // ---- epilogue: normalize rows, write ctx[b, s, h*64 + d] as bf16 ----
#pragma unroll
    for (int v = 0; v < 8; ++v) {
        const float inv = 1.0f / l_run[v];
        const int sRow = q0 + v + g * 8;
        const size_t rowbase = ((size_t)b * SEQ + sRow) * DMODEL + h * HDIM;
#pragma unroll
        for (int nn = 0; nn < 4; ++nn)
            ctx[rowbase + nn * 16 + r] = (__bf16)(o[nn][v] * inv);
    }
}

// ---------------- host launcher ----------------
extern "C" void kernel_launch(void* const* d_in, const int* in_sizes, int n_in,
                              void* d_out, int out_size, void* d_ws, size_t ws_size,
                              hipStream_t stream) {
    const float* x  = (const float*)d_in[0];
    const float* Wq = (const float*)d_in[1];
    const float* Wk = (const float*)d_in[2];
    const float* Wv = (const float*)d_in[3];
    const float* Wo = (const float*)d_in[4];
    const float* bo = (const float*)d_in[5];

    char* ws = (char*)d_ws;
    size_t off = 0;
    auto carve = [&](size_t bytes) -> void* {
        void* p = ws + off;
        off = (off + bytes + 255) & ~(size_t)255;
        return p;
    };
    const size_t xbytes = (size_t)MROWS * DMODEL * sizeof(__bf16);   // 16 MB
    const size_t wbytes = (size_t)DMODEL * DMODEL * sizeof(__bf16);  // 2 MB

    __bf16* x_bf  = (__bf16*)carve(xbytes);
    __bf16* wq_t  = (__bf16*)carve(wbytes);
    __bf16* wk_t  = (__bf16*)carve(wbytes);
    __bf16* wv_t  = (__bf16*)carve(wbytes);
    __bf16* wo_t  = (__bf16*)carve(wbytes);
    __bf16* q_bf  = (__bf16*)carve(xbytes);   // [B,H,S,HD], scaled by QSCALE
    __bf16* k_bf  = (__bf16*)carve(xbytes);   // [B,H,S,HD]
    __bf16* vt_bf = (__bf16*)carve(xbytes);   // [B,H,HD,S]
    __bf16* ctx   = (__bf16*)carve(xbytes);   // [B,S,D]
    (void)ws_size; (void)in_sizes; (void)n_in; (void)out_size;

    // 1) precision conversion / weight transposition (k-major for B-fragments)
    {
        const int n = MROWS * DMODEL;
        cvt_bf16_kernel<<<(n + 255) / 256, 256, 0, stream>>>(x, x_bf, n);
        const int nw = DMODEL * DMODEL;
        transpose_w_kernel<<<(nw + 255) / 256, 256, 0, stream>>>(Wq, wq_t);
        transpose_w_kernel<<<(nw + 255) / 256, 256, 0, stream>>>(Wk, wk_t);
        transpose_w_kernel<<<(nw + 255) / 256, 256, 0, stream>>>(Wv, wv_t);
        transpose_w_kernel<<<(nw + 255) / 256, 256, 0, stream>>>(Wo, wo_t);
    }

    // 2) QKV projections (WMMA GEMMs), writing attention-friendly layouts
    dim3 ggrid(DMODEL / 64, MROWS / 64);   // (16, 128)
    gemm_wmma_kernel<1><<<ggrid, 128, 0, stream>>>(x_bf, wq_t, nullptr, q_bf);
    gemm_wmma_kernel<0><<<ggrid, 128, 0, stream>>>(x_bf, wk_t, nullptr, k_bf);
    gemm_wmma_kernel<2><<<ggrid, 128, 0, stream>>>(x_bf, wv_t, nullptr, vt_bf);

    // 3) causal flash attention
    dim3 fgrid(SEQ / 64, BATCH * NHEAD);   // (32, 64)
    flash_attn_kernel<<<fgrid, 128, 0, stream>>>(q_bf, k_bf, vt_bf, ctx);

    // 4) output projection + bias -> f32
    gemm_wmma_kernel<3><<<ggrid, 128, 0, stream>>>(ctx, wo_t, bo, d_out);
}
